// GFASTKAN_Nodes_49469433315364
// MI455X (gfx1250) — matrix operand order repaired and checked by reading, not compiled, not run on hardware
//
#include <hip/hip_runtime.h>
#include <hip/hip_bf16.h>
#include <cstddef>

#define NN 50000
#define NE 1600000

typedef __bf16 bf16_t;
typedef bf16_t v16bf __attribute__((ext_vector_type(16)));
typedef float  v8f   __attribute__((ext_vector_type(8)));

union Frag { unsigned int u[8]; v16bf v; };

__device__ __forceinline__ unsigned short f2bf(float x) {
    unsigned int b = __float_as_uint(x);
    b += 0x7FFFu + ((b >> 16) & 1u);          // round-to-nearest-even
    return (unsigned short)(b >> 16);
}

__device__ __forceinline__ void atomAddF(float* p, float v) {
    __hip_atomic_fetch_add(p, v, __ATOMIC_RELAXED, __HIP_MEMORY_SCOPE_AGENT);
}

// ---------------- degree / norm ----------------
__global__ void k_deg_init(float* deg, int n) {
    int i = blockIdx.x * blockDim.x + threadIdx.x;
    if (i < n) deg[i] = 1.0f;                 // self loop contributes 1
}
__global__ void k_deg_edges(const int* __restrict__ col, float* deg, int e) {
    int i = blockIdx.x * blockDim.x + threadIdx.x;
    if (i < e) atomAddF(&deg[col[i]], 1.0f);
}
__global__ void k_dis(float* deg, int n) {
    int i = blockIdx.x * blockDim.x + threadIdx.x;
    if (i < n) deg[i] = rsqrtf(deg[i]);       // deg >= 1 always
}

// ---------------- weight conversion: Wb[o][k] bf16, bias = sb+bb ----------------
__global__ void k_convw(const float* __restrict__ sw, const float* __restrict__ bw,
                        const float* __restrict__ sb, const float* __restrict__ bb,
                        unsigned short* __restrict__ Wb, float* __restrict__ biasC,
                        int do_real, int do_pad, int K, int KG, int di) {
    int idx = blockIdx.x * blockDim.x + threadIdx.x;
    if (idx >= do_pad * K) return;
    int o = idx / K, k = idx - o * K;
    float v = 0.f;
    if (o < do_real) v = (k < KG) ? sw[(size_t)o * KG + k] : bw[(size_t)o * di + (k - KG)];
    Wb[idx] = f2bf(v);
    if (k == 0) biasC[o] = (o < do_real) ? (sb[o] + bb[o]) : 0.f;
}

// ---------------- fused FastKAN layer: LN + RBF + SiLU -> WMMA GEMM ----------------
// block = 128 threads (4 waves), 16 nodes per block.
// WMMA phase shares each A-fragment across all TPW column tiles (loop interchange).
template <int DI, int DOPAD, int TPW>
__global__ void k_fkan(const float* __restrict__ in, int ldin,
                       const float* __restrict__ lng, const float* __restrict__ lnb,
                       const unsigned short* __restrict__ Wb,
                       const float* __restrict__ biasC,
                       float* __restrict__ out) {
    constexpr int K = DI * 5;          // di*G spline cols + di silu cols
    constexpr int KG = DI * 4;
    constexpr int ASTR = K + 8;        // LDS row pad
    extern __shared__ char smem[];
    float* hs   = (float*)smem;        // 16*DI
    float* red  = hs + 16 * DI;        // 128
    float* red2 = red + 128;           // 128
    float* mu_s = red2 + 128;          // 16
    float* rs_s = mu_s + 16;           // 16
    unsigned short* As = (unsigned short*)(rs_s + 16);  // 16*ASTR bf16

    const int tid  = threadIdx.x;
    const int base = blockIdx.x * 16;

    for (int idx = tid; idx < 16 * DI; idx += 128) {
        int r = idx / DI, c = idx - r * DI;
        hs[idx] = in[(size_t)(base + r) * ldin + c];
    }
    __syncthreads();

    // LayerNorm stats: 8 threads per row
    const int row = tid >> 3, sub = tid & 7;
    float s = 0.f, s2 = 0.f;
    for (int c = sub; c < DI; c += 8) {
        float h = hs[row * DI + c]; s += h; s2 += h * h;
    }
    red[tid] = s; red2[tid] = s2;
    __syncthreads();
    if (sub == 0) {
        float ts = 0.f, ts2 = 0.f;
        for (int j = 0; j < 8; ++j) { ts += red[row * 8 + j]; ts2 += red2[row * 8 + j]; }
        float mu  = ts * (1.0f / DI);
        float var = ts2 * (1.0f / DI) - mu * mu;
        mu_s[row] = mu;
        rs_s[row] = rsqrtf(var + 1e-5f);
    }
    __syncthreads();

    // Build A = [RBF basis (d*G) | silu(h) (d)] in bf16
    {
        const float mu = mu_s[row], rs = rs_s[row];
        const float invd = 0.75f;  // 1/DENOM, DENOM = 4/3
        for (int c = sub; c < DI; c += 8) {
            float h  = hs[row * DI + c];
            float hn = (h - mu) * rs * lng[c] + lnb[c];
            As[row * ASTR + KG + c] = f2bf(h * (1.f / (1.f + __expf(-h))));
            float t0 = (hn + 2.f)        * invd;
            float t1 = (hn + 2.f / 3.f)  * invd;
            float t2 = (hn - 2.f / 3.f)  * invd;
            float t3 = (hn - 2.f)        * invd;
            As[row * ASTR + c * 4 + 0] = f2bf(__expf(-t0 * t0));
            As[row * ASTR + c * 4 + 1] = f2bf(__expf(-t1 * t1));
            As[row * ASTR + c * 4 + 2] = f2bf(__expf(-t2 * t2));
            As[row * ASTR + c * 4 + 3] = f2bf(__expf(-t3 * t3));
        }
    }
    __syncthreads();

    // WMMA: each wave owns TPW adjacent 16x16 output tiles; A frag shared across tiles
    const int wave = tid >> 5, l = tid & 31;
    const int m = l & 15;
    const int khalfA = (l >> 4) << 3;   // +8  for lanes 16..31 (A frag)
    const int khalfB = (l >> 4) << 3;   // +8  for lanes 16..31 (B frag)
    const int c0base = wave * TPW * 16;
    if (c0base < DOPAD) {
        v8f acc[TPW];
        const unsigned short* Brow[TPW];
#pragma unroll
        for (int tt = 0; tt < TPW; ++tt) {
            int col = c0base + tt * 16 + m;
            float bv = biasC[col];          // C column is lane-uniform -> preload bias
            acc[tt] = (v8f){bv, bv, bv, bv, bv, bv, bv, bv};
            Brow[tt] = Wb + (size_t)col * K + khalfB;
        }
        const unsigned short* Arow = As + m * ASTR + khalfA;
#pragma unroll 4
        for (int kb = 0; kb < K; kb += 32) {
            Frag a;
#pragma unroll
            for (int q = 0; q < 4; ++q)
                a.u[q] = *(const unsigned int*)(Arow + kb + 2 * q);
#pragma unroll
            for (int q = 0; q < 4; ++q)
                a.u[4 + q] = *(const unsigned int*)(Arow + kb + 16 + 2 * q);
#pragma unroll
            for (int tt = 0; tt < TPW; ++tt) {
                Frag b;
#pragma unroll
                for (int q = 0; q < 8; ++q)
                    b.u[q] = *(const unsigned int*)(Brow[tt] + kb + ((q >> 2) << 4) + 2 * (q & 3));
                acc[tt] = __builtin_amdgcn_wmma_f32_16x16x32_bf16(
                    false, a.v, false, b.v, (short)0, acc[tt], false, false);
            }
        }
        int rw = base + ((l >> 4) << 3);
#pragma unroll
        for (int tt = 0; tt < TPW; ++tt) {
            int col = c0base + tt * 16 + m;
#pragma unroll
            for (int r = 0; r < 8; ++r)
                out[(size_t)(rw + r) * DOPAD + col] = acc[tt][r];
        }
    }
}

// ---------------- GCN propagate ----------------
__global__ void k_prop_init(const float* __restrict__ hw, int ldhw,
                            const float* __restrict__ dis, const float* __restrict__ bias,
                            float* __restrict__ out, int ldout, int docols, int n) {
    int idx = blockIdx.x * blockDim.x + threadIdx.x;
    if (idx >= n * docols) return;
    int i = idx / docols, c = idx - i * docols;
    float d = dis[i];
    out[(size_t)i * ldout + c] = d * d * hw[(size_t)i * ldhw + c] + bias[c];
}

__global__ void k_prop_edges128(const int* __restrict__ rowp, const int* __restrict__ colp,
                                const float* __restrict__ dis, const float* __restrict__ hw,
                                float* __restrict__ out, int e) {
    int gw   = (blockIdx.x * blockDim.x + threadIdx.x) >> 5;
    int lane = threadIdx.x & 31;
    if (gw >= e) return;
    int r = rowp[gw], c = colp[gw];
    float nrm = dis[r] * dis[c];
    const float4 v = ((const float4*)(hw + (size_t)r * 128))[lane];
    float* dst = out + (size_t)c * 128 + lane * 4;
    atomAddF(dst + 0, nrm * v.x);
    atomAddF(dst + 1, nrm * v.y);
    atomAddF(dst + 2, nrm * v.z);
    atomAddF(dst + 3, nrm * v.w);
}

__global__ void k_prop_edges40(const int* __restrict__ rowp, const int* __restrict__ colp,
                               const float* __restrict__ dis, const float* __restrict__ hw2,
                               float* __restrict__ out, int e) {
    int gw   = (blockIdx.x * blockDim.x + threadIdx.x) >> 5;
    int lane = threadIdx.x & 31;
    if (gw >= e) return;
    int r = rowp[gw], c = colp[gw];
    float nrm = dis[r] * dis[c];
    const float* src = hw2 + (size_t)r * 48;
    float* dst = out + (size_t)c * 40;
    atomAddF(&dst[lane], nrm * src[lane]);
    if (lane < 8) atomAddF(&dst[32 + lane], nrm * src[32 + lane]);
}

// ---------------- BatchNorm (training-mode batch stats) ----------------
__global__ void k_zero(float* p, int n) {
    int i = blockIdx.x * blockDim.x + threadIdx.x;
    if (i < n) p[i] = 0.f;
}
__global__ void k_bn_stats(const float* __restrict__ h, float* __restrict__ stats, int n) {
    int c = threadIdx.x;                       // 128 threads = 128 cols
    int r0 = blockIdx.x * 256;
    int rend = r0 + 256; if (rend > n) rend = n;
    float s = 0.f, s2 = 0.f;
    for (int r = r0; r < rend; ++r) {
        float v = h[(size_t)r * 128 + c]; s += v; s2 += v * v;
    }
    atomAddF(&stats[c], s);
    atomAddF(&stats[128 + c], s2);
}
__global__ void k_bn_apply(const float* __restrict__ h, const float* __restrict__ stats,
                           const float* __restrict__ g, const float* __restrict__ b,
                           float* __restrict__ xcat, int coloff, int n) {
    int idx = blockIdx.x * blockDim.x + threadIdx.x;
    if (idx >= n * 128) return;
    int i = idx >> 7, c = idx & 127;
    float mean = stats[c] * (1.f / NN);
    float var  = stats[128 + c] * (1.f / NN) - mean * mean;
    float sc   = g[c] * rsqrtf(var + 1e-5f);
    xcat[(size_t)i * 384 + coloff + c] = (h[(size_t)i * 128 + c] - mean) * sc + b[c];
}
__global__ void k_copy_x(const float* __restrict__ x, float* __restrict__ xcat, int n) {
    int idx = blockIdx.x * blockDim.x + threadIdx.x;
    if (idx >= n * 128) return;
    int i = idx >> 7, c = idx & 127;
    xcat[(size_t)i * 384 + c] = x[idx];
}

extern "C" void kernel_launch(void* const* d_in, const int* in_sizes, int n_in,
                              void* d_out, int out_size, void* d_ws, size_t ws_size,
                              hipStream_t stream) {
    (void)in_sizes; (void)n_in; (void)out_size; (void)ws_size;

    const float* x    = (const float*)d_in[0];
    const int*   ei   = (const int*)d_in[1];
    const int*   rowp = ei;
    const int*   colp = ei + NE;
    const float* p[3][7];
    for (int L = 0; L < 3; ++L)
        for (int j = 0; j < 7; ++j) p[L][j] = (const float*)d_in[2 + L * 7 + j];
    const float* bn_g = (const float*)d_in[23];
    const float* bn_b = (const float*)d_in[24];
    float* out = (float*)d_out;

    char* w = (char*)d_ws;
    auto carve = [&](size_t bytes) { char* q = w; w += (bytes + 255) & ~(size_t)255; return q; };
    float* dis  = (float*)carve((size_t)NN * 4);
    float* xcat = (float*)carve((size_t)NN * 384 * 4);
    float* hw   = (float*)carve((size_t)NN * 128 * 4);
    float* hbuf = (float*)carve((size_t)NN * 128 * 4);
    float* hw2  = (float*)carve((size_t)NN * 48 * 4);
    unsigned short* W0 = (unsigned short*)carve((size_t)128 * 640 * 2);
    unsigned short* W1 = (unsigned short*)carve((size_t)128 * 640 * 2);
    unsigned short* W2 = (unsigned short*)carve((size_t)48 * 1920 * 2);
    float* bias0 = (float*)carve(128 * 4);
    float* bias1 = (float*)carve(128 * 4);
    float* bias2 = (float*)carve(48 * 4);
    float* stats = (float*)carve(256 * 4);

    const int shm128 = (16 * 128 + 288) * 4 + 16 * (128 * 5 + 8) * 2;   // ~30 KB
    const int shm384 = (16 * 384 + 288) * 4 + 16 * (384 * 5 + 8) * 2;   // ~87 KB
    hipFuncSetAttribute((const void*)k_fkan<128, 128, 2>,
                        hipFuncAttributeMaxDynamicSharedMemorySize, shm128);
    hipFuncSetAttribute((const void*)k_fkan<384, 48, 1>,
                        hipFuncAttributeMaxDynamicSharedMemorySize, shm384);

    const int TB = 256;
    // gcn_norm
    k_deg_init<<<(NN + TB - 1) / TB, TB, 0, stream>>>(dis, NN);
    k_deg_edges<<<(NE + TB - 1) / TB, TB, 0, stream>>>(colp, dis, NE);
    k_dis<<<(NN + TB - 1) / TB, TB, 0, stream>>>(dis, NN);
    // weights -> bf16 [do][K] + fused bias
    k_convw<<<(128 * 640 + TB - 1) / TB, TB, 0, stream>>>(p[0][2], p[0][4], p[0][3], p[0][5],
                                                          W0, bias0, 128, 128, 640, 512, 128);
    k_convw<<<(128 * 640 + TB - 1) / TB, TB, 0, stream>>>(p[1][2], p[1][4], p[1][3], p[1][5],
                                                          W1, bias1, 128, 128, 640, 512, 128);
    k_convw<<<(48 * 1920 + TB - 1) / TB, TB, 0, stream>>>(p[2][2], p[2][4], p[2][3], p[2][5],
                                                          W2, bias2, 40, 48, 1920, 1536, 384);
    k_copy_x<<<(NN * 128 + TB - 1) / TB, TB, 0, stream>>>(x, xcat, NN);

    const int edgeBlocks = (NE * 32 + TB - 1) / TB;

    // ---- conv0 ----
    k_fkan<128, 128, 2><<<NN / 16, 128, shm128, stream>>>(xcat, 384, p[0][0], p[0][1], W0, bias0, hw);
    k_prop_init<<<(NN * 128 + TB - 1) / TB, TB, 0, stream>>>(hw, 128, dis, p[0][6], hbuf, 128, 128, NN);
    k_prop_edges128<<<edgeBlocks, TB, 0, stream>>>(rowp, colp, dis, hw, hbuf, NE);
    k_zero<<<1, TB, 0, stream>>>(stats, 256);
    k_bn_stats<<<(NN + 255) / 256, 128, 0, stream>>>(hbuf, stats, NN);
    k_bn_apply<<<(NN * 128 + TB - 1) / TB, TB, 0, stream>>>(hbuf, stats, bn_g, bn_b, xcat, 128, NN);

    // ---- conv1 ----
    k_fkan<128, 128, 2><<<NN / 16, 128, shm128, stream>>>(xcat + 128, 384, p[1][0], p[1][1], W1, bias1, hw);
    k_prop_init<<<(NN * 128 + TB - 1) / TB, TB, 0, stream>>>(hw, 128, dis, p[1][6], hbuf, 128, 128, NN);
    k_prop_edges128<<<edgeBlocks, TB, 0, stream>>>(rowp, colp, dis, hw, hbuf, NE);
    k_zero<<<1, TB, 0, stream>>>(stats, 256);
    k_bn_stats<<<(NN + 255) / 256, 128, 0, stream>>>(hbuf, stats, NN);
    k_bn_apply<<<(NN * 128 + TB - 1) / TB, TB, 0, stream>>>(hbuf, stats, bn_g, bn_b, xcat, 256, NN);

    // ---- conv_out on concat [x | h1 | h2] ----
    k_fkan<384, 48, 1><<<NN / 16, 128, shm384, stream>>>(xcat, 384, p[2][0], p[2][1], W2, bias2, hw2);
    k_prop_init<<<(NN * 40 + TB - 1) / TB, TB, 0, stream>>>(hw2, 48, dis, p[2][6], out, 40, 40, NN);
    k_prop_edges40<<<edgeBlocks, TB, 0, stream>>>(rowp, colp, dis, hw2, out, NE);
}
